// LSA_14474039788206
// MI455X (gfx1250) — matrix operand order repaired and checked by reading, not compiled, not run on hardware
//
#include <hip/hip_runtime.h>
#include <hip/hip_bf16.h>
#include <math.h>

// ---------------- problem constants ----------------
#define DIMX   512
#define HEADS  8
#define DHEAD  64
#define BATCH  4
#define SEQ    2048
#define TOK    (BATCH * SEQ)   // 8192 tokens

typedef __attribute__((ext_vector_type(16))) _Float16 v16h;
typedef __attribute__((ext_vector_type(8)))  float    v8f;
typedef unsigned int u32x4 __attribute__((ext_vector_type(4)));
typedef int          i32x8 __attribute__((ext_vector_type(8)));
typedef int          i32x4 __attribute__((ext_vector_type(4)));

#ifndef __has_builtin
#define __has_builtin(x) 0
#endif
#if __has_builtin(__builtin_amdgcn_tensor_load_to_lds) && \
    __has_builtin(__builtin_amdgcn_s_wait_tensorcnt)
#define HAS_TDM 1
#else
#define HAS_TDM 0
#endif

union Frag16 { uint4 u[2]; v16h v; };

__device__ __forceinline__ v8f wmma_f16(v16h a, v16h b, v8f c) {
  // D = A(16x32 f16) * B(32x16 f16) + C(16x16 f32)
  return __builtin_amdgcn_wmma_f32_16x16x32_f16(
      /*neg_a=*/false, a, /*neg_b=*/false, b,
      /*c_mod=*/(short)0, c, /*reuse_a=*/false, /*reuse_b=*/false);
}

#if HAS_TDM
// Issue one 2-D f16 tile load via the Tensor Data Mover (6-arg builtin,
// clang-23 / therock-10.0 flavor).
// pad_interval/pad_amount are the encoded D# fields (ISA 08, group1):
//   LDS gets pad_amount+1 DWORDs of padding every 2<<pad_interval DWORDs.
__device__ __forceinline__ void tdm_load_tile(
    void* lds_ptr, const void* gptr, unsigned pad_interval, unsigned pad_amount,
    unsigned tensor_d0, unsigned tensor_d1, unsigned tile_d0, unsigned tile_d1,
    unsigned stride0) {
  const unsigned lds_addr = (unsigned)(unsigned long long)lds_ptr;  // low 32b = LDS offset
  const unsigned long long ga = (unsigned long long)gptr;
  u32x4 g0;
  g0[0] = 1u;                                    // count=1 (valid user D#)
  g0[1] = lds_addr;                              // lds_addr
  g0[2] = (unsigned)(ga & 0xFFFFFFFFu);          // global_addr[31:0]
  g0[3] = (unsigned)((ga >> 32) & 0x1FFFFFFu)    // global_addr[56:32]
        | (2u << 30);                            // type = 2 ("image")
  i32x8 g1;
  g1[0] = (int)((1u << 16)                       // data_size = 2 bytes
              | (1u << 20)                       // pad_enable
              | (pad_interval << 22) | (pad_amount << 25));
  g1[1] = (int)((tensor_d0 & 0xFFFFu) << 16);                       // dim0[15:0]
  g1[2] = (int)((tensor_d0 >> 16) | ((tensor_d1 & 0xFFFFu) << 16)); // dim0 hi | dim1 lo
  g1[3] = (int)((tensor_d1 >> 16) | (tile_d0 << 16));               // dim1 hi | tile0
  g1[4] = (int)tile_d1;                          // tile1 | tile2=0
  g1[5] = (int)stride0;                          // dim0 stride [31:0]
  g1[6] = 0;                                     // stride hi | dim1 stride lo
  g1[7] = 0;
  const i32x4 z4 = {0, 0, 0, 0};                 // groups 2/3: 2-D tile
  const i32x8 z8 = {0, 0, 0, 0, 0, 0, 0, 0};
  __builtin_amdgcn_tensor_load_to_lds(g0, g1, z4, z4, z8, 0);
}
#endif

// ============================================================
// Kernel 0: weight prep — fp32 [k][n] -> f16 transposed [n][k]
// ============================================================
__global__ __launch_bounds__(256) void lsa_prep_kernel(
    const float* __restrict__ Wq, const float* __restrict__ Wk,
    const float* __restrict__ Wv, const float* __restrict__ Wo,
    _Float16* __restrict__ WT) {   // [4][512n][512k]
  __shared__ __align__(16) _Float16 tile[64][72];
  const float* W = (blockIdx.z == 0) ? Wq
                 : (blockIdx.z == 1) ? Wk
                 : (blockIdx.z == 2) ? Wv : Wo;
  _Float16* dst = WT + (size_t)blockIdx.z * DIMX * DIMX;
  const int k0 = blockIdx.x * 64, n0 = blockIdx.y * 64;
  const int t = threadIdx.x;
  {
    const int r = t >> 2, c0 = (t & 3) * 16;
    const float4* src = (const float4*)(W + (size_t)(k0 + r) * DIMX + n0 + c0);
    #pragma unroll
    for (int q4 = 0; q4 < 4; ++q4) {
      const float4 f = src[q4];
      tile[c0 + q4 * 4 + 0][r] = (_Float16)f.x;
      tile[c0 + q4 * 4 + 1][r] = (_Float16)f.y;
      tile[c0 + q4 * 4 + 2][r] = (_Float16)f.z;
      tile[c0 + q4 * 4 + 3][r] = (_Float16)f.w;
    }
  }
  __syncthreads();
  {
    const int rn = t >> 2, ck = (t & 3) * 16;
    uint4* out = (uint4*)(dst + (size_t)(n0 + rn) * DIMX + k0 + ck);
    out[0] = *(const uint4*)&tile[rn][ck];
    out[1] = *(const uint4*)&tile[rn][ck + 8];
  }
}

// ============================================================
// Kernel 1: LayerNorm (fp32 in -> f16 out), one block per token
// ============================================================
__global__ __launch_bounds__(128) void lsa_ln_kernel(
    const float* __restrict__ x, const float* __restrict__ gamma,
    const float* __restrict__ beta, _Float16* __restrict__ xn) {
  const int row = blockIdx.x;
  const int t   = threadIdx.x;
  __shared__ float red[128];

  const float4 xv = ((const float4*)(x + (size_t)row * DIMX))[t];
  red[t] = xv.x + xv.y + xv.z + xv.w;
  __syncthreads();
  #pragma unroll
  for (int off = 64; off > 0; off >>= 1) {
    if (t < off) red[t] += red[t + off];
    __syncthreads();
  }
  const float mu = red[0] * (1.0f / DIMX);
  __syncthreads();

  const float dx = xv.x - mu, dy = xv.y - mu, dz = xv.z - mu, dw = xv.w - mu;
  red[t] = dx * dx + dy * dy + dz * dz + dw * dw;
  __syncthreads();
  #pragma unroll
  for (int off = 64; off > 0; off >>= 1) {
    if (t < off) red[t] += red[t + off];
    __syncthreads();
  }
  const float rstd = rsqrtf(red[0] * (1.0f / DIMX) + 1e-5f);

  const float4 gv = ((const float4*)gamma)[t];
  const float4 bv = ((const float4*)beta)[t];
  union { _Float16 h[4]; uint2 u; } pk;
  pk.h[0] = (_Float16)(dx * rstd * gv.x + bv.x);
  pk.h[1] = (_Float16)(dy * rstd * gv.y + bv.y);
  pk.h[2] = (_Float16)(dz * rstd * gv.z + bv.z);
  pk.h[3] = (_Float16)(dw * rstd * gv.w + bv.w);
  ((uint2*)(xn + (size_t)row * DIMX))[t] = pk.u;
}

// ============================================================
// Kernel 2: QKV projection GEMM (xn[8192,512] @ W)
// A tile via TDM (fallback: manual), double-buffered LDS.
// B direct from f16 transposed weights (L2 resident).
// ============================================================
__global__ __launch_bounds__(256) void lsa_qkv_kernel(
    const _Float16* __restrict__ xn, const _Float16* __restrict__ WT,
    _Float16* __restrict__ qout, _Float16* __restrict__ kout,
    _Float16* __restrict__ vT) {
  __shared__ __align__(16) _Float16 As[2][128][40];

  const int nBase = blockIdx.x * 256;
  const int mBase = blockIdx.y * 128;
  const _Float16* Wt = WT + (size_t)blockIdx.z * DIMX * DIMX;  // [n][k]

  const int t = threadIdx.x;
  const int wave = t >> 5, lane = t & 31, lr = lane & 15, hi = lane >> 4;
  const int wm = wave >> 2, wn = wave & 3;
  const int k0h = hi ? 8 : 0;
  const int kb  = hi ? 8 : 0;
  const int sr = t >> 1, scol = (t & 1) * 16;   // manual staging coords

  v8f acc[4][4] = {};

  // ---- prologue: k-tile 0 ----
#if HAS_TDM
  if (wave == 0) {
    tdm_load_tile(&As[0][0][0], xn + (size_t)mBase * DIMX,
                  /*pi=16DW*/3, /*pa=4DW*/3, 32, 128, 32, 128, DIMX);
    __builtin_amdgcn_s_wait_tensorcnt((short)0);
  }
#else
  {
    const uint4* src = (const uint4*)(xn + (size_t)(mBase + sr) * DIMX + scol);
    *(uint4*)&As[0][sr][scol]     = src[0];
    *(uint4*)&As[0][sr][scol + 8] = src[1];
  }
#endif
  __syncthreads();

  for (int it = 0; it < DIMX / 32; ++it) {
    const int kt = it * 32;
    const int cur = it & 1;
    if (it + 1 < DIMX / 32) {   // stage next tile into other buffer
#if HAS_TDM
      if (wave == 0)
        tdm_load_tile(&As[cur ^ 1][0][0],
                      xn + (size_t)mBase * DIMX + kt + 32,
                      3, 3, 32, 128, 32, 128, DIMX);
#else
      const uint4* src =
          (const uint4*)(xn + (size_t)(mBase + sr) * DIMX + kt + 32 + scol);
      *(uint4*)&As[cur ^ 1][sr][scol]     = src[0];
      *(uint4*)&As[cur ^ 1][sr][scol + 8] = src[1];
#endif
    }

    Frag16 a[4], b[4];
    #pragma unroll
    for (int mi = 0; mi < 4; ++mi) {
      const _Float16* p = &As[cur][wm * 64 + mi * 16 + lr][k0h];
      a[mi].u[0] = *(const uint4*)p;
      a[mi].u[1] = *(const uint4*)(p + 16);
    }
    #pragma unroll
    for (int ni = 0; ni < 4; ++ni) {
      const _Float16* p =
          Wt + (size_t)(nBase + wn * 64 + ni * 16 + lr) * DIMX + kt + kb;
      b[ni].u[0] = *(const uint4*)p;
      b[ni].u[1] = *(const uint4*)(p + 16);
    }
    #pragma unroll
    for (int mi = 0; mi < 4; ++mi)
      #pragma unroll
      for (int ni = 0; ni < 4; ++ni)
        acc[mi][ni] = wmma_f16(a[mi].v, b[ni].v, acc[mi][ni]);

#if HAS_TDM
    if (wave == 0 && it + 1 < DIMX / 32)
      __builtin_amdgcn_s_wait_tensorcnt((short)0);
#endif
    __syncthreads();
  }

  // ---- epilogue ----
  const bool isV = (blockIdx.z == 2);
  _Float16* outp = (blockIdx.z == 0) ? qout : kout;
  #pragma unroll
  for (int mi = 0; mi < 4; ++mi) {
    const int m0 = mBase + wm * 64 + mi * 16 + hi * 8;
    const int bb = m0 >> 11;
    const int n0 = m0 & (SEQ - 1);
    #pragma unroll
    for (int ni = 0; ni < 4; ++ni) {
      const int c = nBase + wn * 64 + ni * 16 + lr;
      const int h = c >> 6, d = c & 63;
      if (isV) {
        union { _Float16 h8[8]; uint4 u; } pk;
        #pragma unroll
        for (int r = 0; r < 8; ++r) pk.h8[r] = (_Float16)acc[mi][ni][r];
        *(uint4*)&vT[((size_t)(bb * HEADS + h) * DHEAD + d) * SEQ + n0] = pk.u;
      } else {
        #pragma unroll
        for (int r = 0; r < 8; ++r)
          outp[((size_t)((bb * HEADS + h) * SEQ + n0 + r)) * DHEAD + d] =
              (_Float16)acc[mi][ni][r];
      }
    }
  }
}

// ============================================================
// Kernel 3: flash attention; K/V^T tiles via TDM, double-buffered
// ============================================================
__global__ __launch_bounds__(256) void lsa_attn_kernel(
    const _Float16* __restrict__ q, const _Float16* __restrict__ k,
    const _Float16* __restrict__ vT, const float* __restrict__ log_temp,
    _Float16* __restrict__ attOut) {
  __shared__ __align__(16) _Float16 Ks[2][32][72];    // [kv][d]
  __shared__ __align__(16) _Float16 Vt[2][64][40];    // [d][kv]
  __shared__ __align__(16) _Float16 Pb[8][16][40];    // per-wave P bounce

  const int bh = blockIdx.y;
  const int t = threadIdx.x, wave = t >> 5, lane = t & 31;
  const int lr = lane & 15, hi = lane >> 4;
  const int k0h = hi ? 8 : 0, kb = hi ? 8 : 0;
  const float scale = __expf(log_temp[0]);
  const int qRow0 = blockIdx.x * 128 + wave * 16;
  const size_t base = (size_t)bh * SEQ * DHEAD;
  const int skr = t >> 3, skc = (t & 7) * 8;   // manual staging coords
  const int svr = t >> 2, svc = (t & 3) * 8;

  Frag16 qa[2];
  #pragma unroll
  for (int d2 = 0; d2 < 2; ++d2) {
    const _Float16* p = q + base + (size_t)(qRow0 + lr) * DHEAD + d2 * 32 + k0h;
    qa[d2].u[0] = *(const uint4*)p;
    qa[d2].u[1] = *(const uint4*)(p + 16);
  }

  float mrun[8], lrun[8];
  #pragma unroll
  for (int r = 0; r < 8; ++r) { mrun[r] = -3.0e38f; lrun[r] = 0.0f; }
  v8f o[4] = {};

  // ---- prologue: kv tile 0 ----
#if HAS_TDM
  if (wave == 0) {
    tdm_load_tile(&Ks[0][0][0], k + base, /*pi=32DW*/4, /*pa=4DW*/3,
                  64, 32, 64, 32, DHEAD);
    tdm_load_tile(&Vt[0][0][0], vT + base, /*pi=16DW*/3, /*pa=4DW*/3,
                  32, 64, 32, 64, SEQ);
    __builtin_amdgcn_s_wait_tensorcnt((short)0);
  }
#else
  *(uint4*)&Ks[0][skr][skc] =
      *(const uint4*)(k + base + (size_t)skr * DHEAD + skc);
  *(uint4*)&Vt[0][svr][svc] =
      *(const uint4*)(vT + base + (size_t)svr * SEQ + svc);
#endif
  __syncthreads();

  for (int it = 0; it < SEQ / 32; ++it) {
    const int kv0 = it * 32;
    const int cur = it & 1;
    if (it + 1 < SEQ / 32) {   // stage next kv tile
#if HAS_TDM
      if (wave == 0) {
        tdm_load_tile(&Ks[cur ^ 1][0][0],
                      k + base + (size_t)(kv0 + 32) * DHEAD,
                      4, 3, 64, 32, 64, 32, DHEAD);
        tdm_load_tile(&Vt[cur ^ 1][0][0], vT + base + kv0 + 32,
                      3, 3, 32, 64, 32, 64, SEQ);
      }
#else
      *(uint4*)&Ks[cur ^ 1][skr][skc] =
          *(const uint4*)(k + base + (size_t)(kv0 + 32 + skr) * DHEAD + skc);
      *(uint4*)&Vt[cur ^ 1][svr][svc] =
          *(const uint4*)(vT + base + (size_t)svr * SEQ + kv0 + 32 + svc);
#endif
    }

    // ---- S = Q K^T for two 16-col kv subtiles ----
    v8f sc[2];
    #pragma unroll
    for (int s = 0; s < 2; ++s) {
      v8f cacc = {};
      #pragma unroll
      for (int d2 = 0; d2 < 2; ++d2) {
        Frag16 kf;
        const _Float16* p = &Ks[cur][s * 16 + lr][d2 * 32 + kb];
        kf.u[0] = *(const uint4*)p;
        kf.u[1] = *(const uint4*)(p + 16);
        cacc = wmma_f16(qa[d2].v, kf.v, cacc);
      }
      sc[s] = cacc;
    }

    // ---- scale + diagonal mask ----
    #pragma unroll
    for (int s = 0; s < 2; ++s) {
      const int col = kv0 + s * 16 + lr;
      #pragma unroll
      for (int r = 0; r < 8; ++r) {
        const float val = sc[s][r] * scale;
        sc[s][r] = ((qRow0 + r + hi * 8) == col) ? -3.0e38f : val;
      }
    }

    // ---- online softmax over the 32 kv columns ----
    #pragma unroll
    for (int r = 0; r < 8; ++r) {
      float rm = fmaxf(sc[0][r], sc[1][r]);
      #pragma unroll
      for (int m = 1; m < 16; m <<= 1) rm = fmaxf(rm, __shfl_xor(rm, m, 32));
      const float nm = fmaxf(mrun[r], rm);
      const float p0 = __expf(sc[0][r] - nm);
      const float p1 = __expf(sc[1][r] - nm);
      float rs = p0 + p1;
      #pragma unroll
      for (int m = 1; m < 16; m <<= 1) rs += __shfl_xor(rs, m, 32);
      const float corr = __expf(mrun[r] - nm);
      lrun[r] = lrun[r] * corr + rs;
      mrun[r] = nm;
      #pragma unroll
      for (int j = 0; j < 4; ++j) o[j][r] *= corr;
      Pb[wave][r + hi * 8][lr]      = (_Float16)p0;
      Pb[wave][r + hi * 8][16 + lr] = (_Float16)p1;
    }
    __builtin_amdgcn_wave_barrier();   // DS is in-order per wave

    // ---- re-layout P and O += P V ----
    Frag16 pa;
    {
      const _Float16* p = &Pb[wave][lr][k0h];
      pa.u[0] = *(const uint4*)p;
      pa.u[1] = *(const uint4*)(p + 16);
    }
    #pragma unroll
    for (int j = 0; j < 4; ++j) {
      Frag16 vf;
      const _Float16* p = &Vt[cur][j * 16 + lr][kb];
      vf.u[0] = *(const uint4*)p;
      vf.u[1] = *(const uint4*)(p + 16);
      o[j] = wmma_f16(pa.v, vf.v, o[j]);
    }

#if HAS_TDM
    if (wave == 0 && it + 1 < SEQ / 32)
      __builtin_amdgcn_s_wait_tensorcnt((short)0);
#endif
    __syncthreads();
  }

  // ---- normalize and store [b, n, h*64+d] f16 ----
  const int bb = bh >> 3, hh = bh & 7;
  #pragma unroll
  for (int r = 0; r < 8; ++r) {
    const float inv = 1.0f / lrun[r];
    const int n = qRow0 + r + hi * 8;
    const size_t row = (size_t)(bb * SEQ + n) * DIMX + hh * DHEAD;
    #pragma unroll
    for (int j = 0; j < 4; ++j)
      attOut[row + j * 16 + lr] = (_Float16)(o[j][r] * inv);
  }
}

// ============================================================
// Kernel 4: output projection (att[8192,512] @ Wo + bo) -> fp32
// ============================================================
__global__ __launch_bounds__(256) void lsa_out_kernel(
    const _Float16* __restrict__ A, const _Float16* __restrict__ WoT,
    const float* __restrict__ bo, float* __restrict__ out) {
  __shared__ __align__(16) _Float16 As[2][128][40];

  const int nBase = blockIdx.x * 256;
  const int mBase = blockIdx.y * 128;
  const int t = threadIdx.x;
  const int wave = t >> 5, lane = t & 31, lr = lane & 15, hi = lane >> 4;
  const int wm = wave >> 2, wn = wave & 3;
  const int k0h = hi ? 8 : 0, kb = hi ? 8 : 0;
  const int sr = t >> 1, scol = (t & 1) * 16;

  v8f acc[4][4] = {};

#if HAS_TDM
  if (wave == 0) {
    tdm_load_tile(&As[0][0][0], A + (size_t)mBase * DIMX,
                  3, 3, 32, 128, 32, 128, DIMX);
    __builtin_amdgcn_s_wait_tensorcnt((short)0);
  }
#else
  {
    const uint4* src = (const uint4*)(A + (size_t)(mBase + sr) * DIMX + scol);
    *(uint4*)&As[0][sr][scol]     = src[0];
    *(uint4*)&As[0][sr][scol + 8] = src[1];
  }
#endif
  __syncthreads();

  for (int it = 0; it < DIMX / 32; ++it) {
    const int kt = it * 32;
    const int cur = it & 1;
    if (it + 1 < DIMX / 32) {
#if HAS_TDM
      if (wave == 0)
        tdm_load_tile(&As[cur ^ 1][0][0],
                      A + (size_t)mBase * DIMX + kt + 32,
                      3, 3, 32, 128, 32, 128, DIMX);
#else
      const uint4* src =
          (const uint4*)(A + (size_t)(mBase + sr) * DIMX + kt + 32 + scol);
      *(uint4*)&As[cur ^ 1][sr][scol]     = src[0];
      *(uint4*)&As[cur ^ 1][sr][scol + 8] = src[1];
#endif
    }

    Frag16 a[4], b[4];
    #pragma unroll
    for (int mi = 0; mi < 4; ++mi) {
      const _Float16* p = &As[cur][wm * 64 + mi * 16 + lr][k0h];
      a[mi].u[0] = *(const uint4*)p;
      a[mi].u[1] = *(const uint4*)(p + 16);
    }
    #pragma unroll
    for (int ni = 0; ni < 4; ++ni) {
      const _Float16* p =
          WoT + (size_t)(nBase + wn * 64 + ni * 16 + lr) * DIMX + kt + kb;
      b[ni].u[0] = *(const uint4*)p;
      b[ni].u[1] = *(const uint4*)(p + 16);
    }
    #pragma unroll
    for (int mi = 0; mi < 4; ++mi)
      #pragma unroll
      for (int ni = 0; ni < 4; ++ni)
        acc[mi][ni] = wmma_f16(a[mi].v, b[ni].v, acc[mi][ni]);

#if HAS_TDM
    if (wave == 0 && it + 1 < DIMX / 32)
      __builtin_amdgcn_s_wait_tensorcnt((short)0);
#endif
    __syncthreads();
  }

  #pragma unroll
  for (int mi = 0; mi < 4; ++mi) {
    const int m0 = mBase + wm * 64 + mi * 16 + hi * 8;
    #pragma unroll
    for (int ni = 0; ni < 4; ++ni) {
      const int c = nBase + wn * 64 + ni * 16 + lr;
      const float bias = bo[c];
      #pragma unroll
      for (int r = 0; r < 8; ++r)
        out[(size_t)(m0 + r) * DIMX + c] = acc[mi][ni][r] + bias;
    }
  }
}

// ============================================================
extern "C" void kernel_launch(void* const* d_in, const int* in_sizes, int n_in,
                              void* d_out, int out_size, void* d_ws,
                              size_t ws_size, hipStream_t stream) {
  const float* x     = (const float*)d_in[0];
  const float* gamma = (const float*)d_in[1];
  const float* beta  = (const float*)d_in[2];
  const float* Wq    = (const float*)d_in[3];
  const float* Wk    = (const float*)d_in[4];
  const float* Wv    = (const float*)d_in[5];
  const float* Wo    = (const float*)d_in[6];
  const float* bo    = (const float*)d_in[7];
  const float* ltmp  = (const float*)d_in[8];
  float* out = (float*)d_out;

  // workspace (f16): xn | Q | K | V^T | WT(4 matrices)
  const size_t SEG  = (size_t)TOK * DIMX * sizeof(_Float16);    // 8 MB
  const size_t WSEG = (size_t)DIMX * DIMX * sizeof(_Float16);   // 512 KB
  char* ws = (char*)d_ws;
  _Float16* xn  = (_Float16*)(ws);            // also attn output buffer
  _Float16* qb  = (_Float16*)(ws + 1 * SEG);
  _Float16* kb  = (_Float16*)(ws + 2 * SEG);
  _Float16* vb  = (_Float16*)(ws + 3 * SEG);  // transposed [b,h,d,n]
  _Float16* WT  = (_Float16*)(ws + 4 * SEG);  // [4][512][512] f16, [n][k]
  _Float16* att = xn;

  lsa_prep_kernel<<<dim3(8, 8, 4), 256, 0, stream>>>(Wq, Wk, Wv, Wo, WT);
  lsa_ln_kernel<<<TOK, 128, 0, stream>>>(x, gamma, beta, xn);
  lsa_qkv_kernel<<<dim3(DIMX / 256, TOK / 128, 3), 256, 0, stream>>>(
      xn, WT, qb, kb, vb);
  lsa_attn_kernel<<<dim3(SEQ / 128, BATCH * HEADS), 256, 0, stream>>>(
      qb, kb, vb, ltmp, att);
  lsa_out_kernel<<<dim3(DIMX / 256, TOK / 128), 256, 0, stream>>>(
      att, WT + 3 * (WSEG / sizeof(_Float16)), bo, out);
}
